// CausalSelfAttention_39049842655599
// MI455X (gfx1250) — matrix operand order repaired and checked by reading, not compiled, not run on hardware
//
#include <hip/hip_runtime.h>

typedef __attribute__((ext_vector_type(16))) __bf16 v16bf;
typedef __attribute__((ext_vector_type(8)))  __bf16 v8bf;
typedef __attribute__((ext_vector_type(4)))  __bf16 v4bf;
typedef __attribute__((ext_vector_type(8)))  float  v8f;

union V16 { v16bf v; v8bf h[2]; };

#define TLEN   2048
#define DMODEL 2048
#define NHEAD  16
#define HDIM   128

static __device__ __forceinline__ v8f wmma_bf16(const V16& a, const V16& b, v8f c) {
    return __builtin_amdgcn_wmma_f32_16x16x32_bf16(false, a.v, false, b.v,
                                                   (short)0, c, false, false);
}

// CDNA5 async global->LDS copy (ASYNCcnt), 16B per lane.
// LDS destination is the low 32 bits of the flat pointer (LDS aperture layout).
static __device__ __forceinline__ void async_ld_b128(const __bf16* lds_dst,
                                                     const __bf16* gsrc) {
    unsigned loff = (unsigned)(size_t)lds_dst;
    asm volatile("global_load_async_to_lds_b128 %0, %1, off"
                 :: "v"(loff), "v"(gsrc)
                 : "memory");
}
static __device__ __forceinline__ void wait_async0() {
    asm volatile("s_wait_asynccnt 0" ::: "memory");
}

// ---------------------------------------------------------------------------
// Kernel 0: fp32 -> bf16 conversion, 4 elements/thread
// ---------------------------------------------------------------------------
__global__ void cvt_f32_bf16(const float* __restrict__ src, __bf16* __restrict__ dst) {
    int i = blockIdx.x * blockDim.x + threadIdx.x;
    float4 f = ((const float4*)src)[i];
    v4bf o;
    o[0] = (__bf16)f.x; o[1] = (__bf16)f.y; o[2] = (__bf16)f.z; o[3] = (__bf16)f.w;
    ((v4bf*)dst)[i] = o;
}

// ---------------------------------------------------------------------------
// Kernel 1: fused QKV projection + RoPE.
// grid = (T/128, 3*NHEAD); block = 256 (8 waves). Each block computes a
// 128(T) x 128(one head) tile of Q, K or V; RoPE fused in registers for Q/K.
// Double-buffered LDS staging via async global->LDS copies.
// Output layout: [H][T][128] bf16.
// ---------------------------------------------------------------------------
__global__ __launch_bounds__(256) void qkv_rope_gemm(
        const __bf16* __restrict__ xb,
        const __bf16* __restrict__ Wqb,
        const __bf16* __restrict__ Wkb,
        const __bf16* __restrict__ Wvb,
        __bf16* __restrict__ qws,
        __bf16* __restrict__ kws,
        __bf16* __restrict__ vws) {
    __shared__ __align__(16) __bf16 lA[2][128 * 32];
    __shared__ __align__(16) __bf16 lB[2][128 * 32];

    const int wave = threadIdx.x >> 5;
    const int lane = threadIdx.x & 31;
    const int half = lane >> 4;
    const int lm   = lane & 15;

    const int t0  = blockIdx.x * 128;
    const int mat = blockIdx.y >> 4;   // 0=Q 1=K 2=V
    const int hh  = blockIdx.y & 15;
    const int o0  = hh * HDIM;

    const __bf16* W  = (mat == 0) ? Wqb : (mat == 1) ? Wkb : Wvb;
    __bf16*      dst = (mat == 0) ? qws : (mat == 1) ? kws : vws;

    auto issue_tile = [&](int buf, int kk) {
#pragma unroll
        for (int i = 0; i < 2; ++i) {
            int c = threadIdx.x + i * 256;      // 0..511
            int row = c >> 2, part = c & 3;     // 128 rows x 4 x 16B
            async_ld_b128(&lA[buf][row * 32 + part * 8],
                          &xb[(size_t)(t0 + row) * DMODEL + kk + part * 8]);
            async_ld_b128(&lB[buf][row * 32 + part * 8],
                          &W[(size_t)(o0 + row) * DMODEL + kk + part * 8]);
        }
    };

    v8f acc[8];
#pragma unroll
    for (int i = 0; i < 8; ++i)
#pragma unroll
        for (int e = 0; e < 8; ++e) acc[i][e] = 0.0f;

    issue_tile(0, 0);
    int cur = 0;
    for (int kk = 0; kk < DMODEL; kk += 32) {
        wait_async0();          // my async stores to buf[cur] landed in LDS
        __syncthreads();        // everyone's landed; prior reads of buf[cur^1] done
        if (kk + 32 < DMODEL) issue_tile(cur ^ 1, kk + 32);

        V16 av;
        av.h[0] = *(const v8bf*)&lA[cur][(wave * 16 + lm) * 32 + half * 8];
        av.h[1] = *(const v8bf*)&lA[cur][(wave * 16 + lm) * 32 + 16 + half * 8];
#pragma unroll
        for (int nt = 0; nt < 8; ++nt) {
            V16 bv;
            bv.h[0] = *(const v8bf*)&lB[cur][(nt * 16 + lm) * 32 + half * 8];
            bv.h[1] = *(const v8bf*)&lB[cur][(nt * 16 + lm) * 32 + 16 + half * 8];
            acc[nt] = wmma_bf16(av, bv, acc[nt]);
        }
        cur ^= 1;
    }

    if (mat == 2) {
#pragma unroll
        for (int nt = 0; nt < 8; ++nt)
#pragma unroll
            for (int r = 0; r < 8; ++r) {
                int t = t0 + wave * 16 + r + 8 * half;
                int d = nt * 16 + lm;
                dst[((size_t)hh * TLEN + t) * HDIM + d] = (__bf16)acc[nt][r];
            }
    } else {
        // RoPE: column pairs (cf, cf+64) live in acc[nt] / acc[nt+4], same lane/elem
#pragma unroll
        for (int nt = 0; nt < 4; ++nt)
#pragma unroll
            for (int r = 0; r < 8; ++r) {
                int t  = t0 + wave * 16 + r + 8 * half;
                int cf = nt * 16 + lm;              // 0..63
                float x1 = acc[nt][r], x2 = acc[nt + 4][r];
                float cth = 1.0f, sth = 0.0f;
                if (cf < 32) {
                    float af = exp2f(-10.0f * (float)cf / 31.0f);
                    float th = (float)t * af;
                    cth = cosf(th);
                    sth = sinf(th);
                }
                float y1 =  x1 * cth + x2 * sth;
                float y2 = -x1 * sth + x2 * cth;
                size_t base = ((size_t)hh * TLEN + t) * HDIM;
                dst[base + cf]      = (__bf16)y1;
                dst[base + cf + 64] = (__bf16)y2;
            }
    }
}

// ---------------------------------------------------------------------------
// Kernel 2: causal flash attention per (head, 128-query tile).
// grid = (T/128, NHEAD); block = 256 (8 waves; wave w owns query rows w*16..+15)
// K tile staged via async global->LDS; V staged transposed via VGPR scatter.
// Writes y/3 in bf16 laid out [T][H*128] for the output projection.
// ---------------------------------------------------------------------------
__global__ __launch_bounds__(256) void attention_fwd(
        const __bf16* __restrict__ qg,
        const __bf16* __restrict__ kg,
        const __bf16* __restrict__ vg,
        __bf16* __restrict__ yb) {
    __shared__ __align__(16) __bf16 lK[32 * 128];        // [s][d]
    __shared__ __align__(16) __bf16 lVT[128 * 32];       // [d][s]
    __shared__ __align__(16) __bf16 lP[8 * 16 * 32];     // per-wave P staging

    const int wave = threadIdx.x >> 5;
    const int lane = threadIdx.x & 31;
    const int half = lane >> 4;
    const int lm   = lane & 15;

    const int hh = blockIdx.y;
    const int t0 = blockIdx.x * 128;
    const int m0 = wave * 16;
    const size_t hb = (size_t)hh * TLEN * HDIM;

    // Q tile (16 rows x 128) into A-layout registers, 4 K-steps of 32
    V16 qa[4];
    {
        const __bf16* qrow = qg + hb + (size_t)(t0 + m0 + lm) * HDIM;
#pragma unroll
        for (int ks = 0; ks < 4; ++ks) {
            qa[ks].h[0] = *(const v8bf*)&qrow[ks * 32 + half * 8];
            qa[ks].h[1] = *(const v8bf*)&qrow[ks * 32 + 16 + half * 8];
        }
    }

    v8f o_acc[8];
#pragma unroll
    for (int i = 0; i < 8; ++i)
#pragma unroll
        for (int e = 0; e < 8; ++e) o_acc[i][e] = 0.0f;

    float m_run[8], l_run[8];
#pragma unroll
    for (int r = 0; r < 8; ++r) { m_run[r] = -3.0e38f; l_run[r] = 0.0f; }

    const float sscale = 0.0883883476483184f;   // sqrt(128)/128
    const int nblk = (t0 + 128) >> 5;           // causal: only key blocks <= tile end
    __bf16* lPw = lP + wave * 16 * 32;

    for (int b = 0; b < nblk; ++b) {
        const int s0 = b * 32;
        // stage K [s][d] (async) and V transposed [d][s] (scatter)
#pragma unroll
        for (int i = 0; i < 2; ++i) {
            int c = threadIdx.x + i * 256;      // 0..511
            int row = c >> 4, part = c & 15;    // 32 rows x 16 x 16B
            async_ld_b128(&lK[row * 128 + part * 8],
                          &kg[hb + (size_t)(s0 + row) * HDIM + part * 8]);
            v8bf vv = *(const v8bf*)&vg[hb + (size_t)(s0 + row) * HDIM + part * 8];
#pragma unroll
            for (int e = 0; e < 8; ++e) lVT[(part * 8 + e) * 32 + row] = vv[e];
        }
        wait_async0();
        __syncthreads();

        // S = Q @ K^T : two 16-wide key tiles
        v8f sa0, sa1;
#pragma unroll
        for (int e = 0; e < 8; ++e) { sa0[e] = 0.0f; sa1[e] = 0.0f; }
#pragma unroll
        for (int ks = 0; ks < 4; ++ks) {
            V16 bv;
            bv.h[0] = *(const v8bf*)&lK[lm * 128 + ks * 32 + half * 8];
            bv.h[1] = *(const v8bf*)&lK[lm * 128 + ks * 32 + 16 + half * 8];
            sa0 = wmma_bf16(qa[ks], bv, sa0);
            bv.h[0] = *(const v8bf*)&lK[(16 + lm) * 128 + ks * 32 + half * 8];
            bv.h[1] = *(const v8bf*)&lK[(16 + lm) * 128 + ks * 32 + 16 + half * 8];
            sa1 = wmma_bf16(qa[ks], bv, sa1);
        }

        // causal mask + online softmax (row = elem r + 8*half, cols across 16-lane group)
#pragma unroll
        for (int r = 0; r < 8; ++r) {
            int t = t0 + m0 + r + 8 * half;
            float sv0 = sa0[r] * sscale;
            float sv1 = sa1[r] * sscale;
            if (s0 + lm > t)      sv0 = -1.0e30f;
            if (s0 + 16 + lm > t) sv1 = -1.0e30f;
            float rmax = fmaxf(sv0, sv1);
#pragma unroll
            for (int mm = 8; mm >= 1; mm >>= 1)
                rmax = fmaxf(rmax, __shfl_xor(rmax, mm, 32));
            float mnew = fmaxf(m_run[r], rmax);
            float sc = __expf(m_run[r] - mnew);
            float e0 = __expf(sv0 - mnew);
            float e1 = __expf(sv1 - mnew);
            float ps = e0 + e1;
#pragma unroll
            for (int mm = 8; mm >= 1; mm >>= 1)
                ps += __shfl_xor(ps, mm, 32);
            l_run[r] = l_run[r] * sc + ps;
            m_run[r] = mnew;
#pragma unroll
            for (int nt = 0; nt < 8; ++nt) o_acc[nt][r] *= sc;
            lPw[(r + 8 * half) * 32 + lm]      = (__bf16)e0;
            lPw[(r + 8 * half) * 32 + 16 + lm] = (__bf16)e1;
        }
        asm volatile("s_wait_dscnt 0" ::: "memory");   // P LDS write -> read (same wave)

        // O += P @ V
        V16 pa;
        pa.h[0] = *(const v8bf*)&lPw[lm * 32 + half * 8];
        pa.h[1] = *(const v8bf*)&lPw[lm * 32 + 16 + half * 8];
#pragma unroll
        for (int nt = 0; nt < 8; ++nt) {
            V16 bv;
            bv.h[0] = *(const v8bf*)&lVT[(nt * 16 + lm) * 32 + half * 8];
            bv.h[1] = *(const v8bf*)&lVT[(nt * 16 + lm) * 32 + 16 + half * 8];
            o_acc[nt] = wmma_bf16(pa, bv, o_acc[nt]);
        }
        __syncthreads();
    }

    // epilogue: 1/l softmax normalization and the reference's /3, bf16 out
#pragma unroll
    for (int r = 0; r < 8; ++r) {
        float inv = 1.0f / (3.0f * l_run[r]);
        int t = t0 + m0 + r + 8 * half;
        size_t base = (size_t)t * DMODEL + hh * HDIM;
#pragma unroll
        for (int nt = 0; nt < 8; ++nt)
            yb[base + nt * 16 + lm] = (__bf16)(o_acc[nt][r] * inv);
    }
}

// ---------------------------------------------------------------------------
// Kernel 3: output projection  out = Y @ Wo^T  (fp32 out)
// grid = (T/128, D/128); block = 256; async double-buffered staging
// ---------------------------------------------------------------------------
__global__ __launch_bounds__(256) void out_proj(
        const __bf16* __restrict__ yb,
        const __bf16* __restrict__ Wob,
        float* __restrict__ out) {
    __shared__ __align__(16) __bf16 lA[2][128 * 32];
    __shared__ __align__(16) __bf16 lB[2][128 * 32];

    const int wave = threadIdx.x >> 5;
    const int lane = threadIdx.x & 31;
    const int half = lane >> 4;
    const int lm   = lane & 15;

    const int t0 = blockIdx.x * 128;
    const int o0 = blockIdx.y * 128;

    auto issue_tile = [&](int buf, int kk) {
#pragma unroll
        for (int i = 0; i < 2; ++i) {
            int c = threadIdx.x + i * 256;
            int row = c >> 2, part = c & 3;
            async_ld_b128(&lA[buf][row * 32 + part * 8],
                          &yb[(size_t)(t0 + row) * DMODEL + kk + part * 8]);
            async_ld_b128(&lB[buf][row * 32 + part * 8],
                          &Wob[(size_t)(o0 + row) * DMODEL + kk + part * 8]);
        }
    };

    v8f acc[8];
#pragma unroll
    for (int i = 0; i < 8; ++i)
#pragma unroll
        for (int e = 0; e < 8; ++e) acc[i][e] = 0.0f;

    issue_tile(0, 0);
    int cur = 0;
    for (int kk = 0; kk < DMODEL; kk += 32) {
        wait_async0();
        __syncthreads();
        if (kk + 32 < DMODEL) issue_tile(cur ^ 1, kk + 32);

        V16 av;
        av.h[0] = *(const v8bf*)&lA[cur][(wave * 16 + lm) * 32 + half * 8];
        av.h[1] = *(const v8bf*)&lA[cur][(wave * 16 + lm) * 32 + 16 + half * 8];
#pragma unroll
        for (int nt = 0; nt < 8; ++nt) {
            V16 bv;
            bv.h[0] = *(const v8bf*)&lB[cur][(nt * 16 + lm) * 32 + half * 8];
            bv.h[1] = *(const v8bf*)&lB[cur][(nt * 16 + lm) * 32 + 16 + half * 8];
            acc[nt] = wmma_bf16(av, bv, acc[nt]);
        }
        cur ^= 1;
    }

#pragma unroll
    for (int nt = 0; nt < 8; ++nt)
#pragma unroll
        for (int r = 0; r < 8; ++r) {
            int t = t0 + wave * 16 + r + 8 * half;
            out[(size_t)t * DMODEL + o0 + nt * 16 + lm] = acc[nt][r];
        }
}

// ---------------------------------------------------------------------------
extern "C" void kernel_launch(void* const* d_in, const int* in_sizes, int n_in,
                              void* d_out, int out_size, void* d_ws, size_t ws_size,
                              hipStream_t stream) {
    const float* x  = (const float*)d_in[0];
    const float* Wq = (const float*)d_in[1];
    const float* Wk = (const float*)d_in[2];
    const float* Wv = (const float*)d_in[3];
    const float* Wo = (const float*)d_in[4];

    const size_t NEL = (size_t)DMODEL * TLEN;   // 4M elements per matrix
    const size_t BYT = NEL * sizeof(__bf16);    // 8 MiB per bf16 matrix
    char* ws = (char*)d_ws;
    __bf16* xb   = (__bf16*)(ws + 0 * BYT);
    __bf16* Wqb  = (__bf16*)(ws + 1 * BYT);
    __bf16* Wkb  = (__bf16*)(ws + 2 * BYT);
    __bf16* Wvb  = (__bf16*)(ws + 3 * BYT);
    __bf16* Wob  = (__bf16*)(ws + 4 * BYT);
    __bf16* qws  = (__bf16*)(ws + 5 * BYT);
    __bf16* kws  = (__bf16*)(ws + 6 * BYT);
    __bf16* vws  = (__bf16*)(ws + 7 * BYT);
    __bf16* ybuf = (__bf16*)(ws + 8 * BYT);

    const int cvtBlocks = (int)(NEL / 4 / 256);
    cvt_f32_bf16<<<cvtBlocks, 256, 0, stream>>>(x,  xb);
    cvt_f32_bf16<<<cvtBlocks, 256, 0, stream>>>(Wq, Wqb);
    cvt_f32_bf16<<<cvtBlocks, 256, 0, stream>>>(Wk, Wkb);
    cvt_f32_bf16<<<cvtBlocks, 256, 0, stream>>>(Wv, Wvb);
    cvt_f32_bf16<<<cvtBlocks, 256, 0, stream>>>(Wo, Wob);

    qkv_rope_gemm<<<dim3(TLEN / 128, 3 * NHEAD), 256, 0, stream>>>(
        xb, Wqb, Wkb, Wvb, qws, kws, vws);

    attention_fwd<<<dim3(TLEN / 128, NHEAD), 256, 0, stream>>>(
        qws, kws, vws, ybuf);

    out_proj<<<dim3(TLEN / 128, DMODEL / 128), 256, 0, stream>>>(
        ybuf, Wob, (float*)d_out);
}
